// privacyLoss2_8203387535488
// MI455X (gfx1250) — compile-verified, matrix operand order checked by reading, not run on hardware
//
#include <hip/hip_runtime.h>
#include <hip/hip_bf16.h>
#include <math.h>

// ---------------------------------------------------------------------------
// privacy loss on MI455X (gfx1250, wave32, WMMA)
//
// Sizes: B=16384 rows, K=128 features, C=32 classes.
// Heavy op: per-class Gram R_c = sum_{i in c} f_i f_i^T  (537 MFLOP)
//   -> V_WMMA_F32_16X16X32_F16, f16 operands / f32 accumulate.
// S_c = R_c/n_c - mu_c mu_c^T + I  (algebraic re-centering, keeps the hot
// loop mask-multiply only).
//
// LDS staging is feature-major so every WMMA fragment is two contiguous
// 16-byte runs -> ds_load_b128, and staging writes are ds_store_b128.
// Double-buffered so global traffic overlaps the matrix pipe.
// ---------------------------------------------------------------------------

typedef __attribute__((ext_vector_type(16))) _Float16 v16h;
typedef __attribute__((ext_vector_type(8)))  _Float16 v8h;
typedef __attribute__((ext_vector_type(8)))  float    v8f;

#define NB 16384
#define NK 128
#define NC 32
#define LT 40            // LDS row pitch in halfs (32 batch + 8 pad) = 80 B

// ---- workspace layout (in floats) -----------------------------------------
#define OFF_MUSUM  0           // [C*K]   = 4096 f32
#define OFF_COUNTS 4096        // [C]     int32
#define OFF_FIRST  4128        // [C]     int32
#define OFF_MU     4160        // [C*K]   = 4096 f32
#define OFF_LD2    8256        // [C]     f32
#define OFF_KL     8288        // [C*C]   = 1024 f32
#define OFF_S      9312        // [C*K*K] = 524288 f32 (16B aligned)
#define OFF_INV    (OFF_S   + NC*NK*NK)   // 524288 f32
#define OFF_AUG    (OFF_INV + NC*NK*NK)   // [C*128*256] = 1048576 f32
// total ~= 2,106,464 floats ~= 8.43 MB of d_ws

// ---------------------------------------------------------------------------
__global__ void k_init(float* musum, int* counts, int* first) {
    int i = blockIdx.x * blockDim.x + threadIdx.x;
    if (i < NC * NK) musum[i] = 0.0f;
    if (i < NC) { counts[i] = 0; first[i] = NB; }
}

// per-class feature sums + class counts; 64 blocks x 128 threads, 256 rows/blk
__global__ void k_classsum(const float* __restrict__ feat,
                           const int*   __restrict__ lab,
                           float* musum, int* counts) {
    __shared__ float acc[NC][NK];
    __shared__ int   lcnt[NC];
    int t = threadIdx.x; // 0..127, column index
    for (int c = 0; c < NC; ++c) acc[c][t] = 0.0f;
    if (t < NC) lcnt[t] = 0;
    __syncthreads();
    int row0 = blockIdx.x * 256;
    for (int r = 0; r < 256; ++r) {
        int lb = lab[row0 + r];
        acc[lb][t] += feat[(size_t)(row0 + r) * NK + t];
    }
    for (int r = t; r < 256; r += 128) atomicAdd(&lcnt[lab[row0 + r]], 1);
    __syncthreads();
    for (int c = 0; c < NC; ++c) atomicAdd(&musum[c * NK + t], acc[c][t]);
    if (t < NC) atomicAdd(&counts[t], lcnt[t]);
}

__global__ void k_first(const int* __restrict__ lab, int* first) {
    int b = blockIdx.x * blockDim.x + threadIdx.x;
    if (b < NB) atomicMin(&first[lab[b]], b);
}

__global__ void k_mu(const float* __restrict__ musum,
                     const int*   __restrict__ counts, float* mu) {
    int i = blockIdx.x * blockDim.x + threadIdx.x;
    if (i < NC * NK) {
        int c = i >> 7;
        int n = counts[c];
        mu[i] = (n > 0) ? musum[i] / (float)n : 0.0f;
    }
}

// ---------------------------------------------------------------------------
// Masked per-class Gram via WMMA. grid = 32 (one class/block), block = 256
// (8 waves). Wave w owns output rows [16w, 16w+16) as 8 tiles of 16x16.
// ---------------------------------------------------------------------------

// Stage a 32-row x 128-col chunk of masked features, feature-major, f16.
// Thread t owns 2 feature columns x 8 batch rows:
//   group g = t>>6 -> batch rows [8g, 8g+8); ci = t&63 -> cols {2ci, 2ci+1}.
// Global reads: 8 coalesced float2 loads (a wave covers 256 B per row).
// LDS writes: one ds_store_b128 per column (8 halfs contiguous along k).
__device__ __forceinline__ void stage_chunk(const float* __restrict__ feat,
                                            const int*   __restrict__ lab,
                                            int c, int base,
                                            _Float16* __restrict__ buf, int t) {
    const int r0   = (t >> 6) * 8;
    const int ci   = t & 63;
    const int col0 = ci * 2;
    v8h a0, a1;
#pragma unroll
    for (int i = 0; i < 8; ++i) {
        int row = base + r0 + i;
        float2 v = *(const float2*)(feat + (size_t)row * NK + col0);
        float  m = (lab[row] == c) ? 1.0f : 0.0f;
        a0[i] = (_Float16)(v.x * m);
        a1[i] = (_Float16)(v.y * m);
        // prefetch two chunks ahead (global_prefetch_b8); 1-in-16 threads
        if ((ci & 15) == 0 && base + 64 < NB)
            __builtin_prefetch(feat + (size_t)(row + 64) * NK + col0, 0, 1);
    }
    *(v8h*)(buf + (size_t)col0 * LT + r0)       = a0;
    *(v8h*)(buf + (size_t)(col0 + 1) * LT + r0) = a1;
}

// Fragment for output tile column `col` (a feature index strip):
// lane's 16 halfs = X[col][k] for k in {hiK..hiK+7} U {16+hiK..16+hiK+7}
// -> two 16-byte contiguous LDS loads (ds_load_b128).
__device__ __forceinline__ v16h frag(const _Float16* __restrict__ buf,
                                     int col, int hiK) {
    const _Float16* p = buf + (size_t)col * LT;
    v8h lo = *(const v8h*)(p + hiK);
    v8h hi = *(const v8h*)(p + 16 + hiK);
    return __builtin_shufflevector(lo, hi, 0, 1, 2, 3, 4, 5, 6, 7,
                                           8, 9, 10, 11, 12, 13, 14, 15);
}

__global__ void k_gram(const float* __restrict__ feat,
                       const int*   __restrict__ lab,
                       const float* __restrict__ mu,
                       const int*   __restrict__ counts,
                       float* __restrict__ S) {
    __shared__ __align__(16) _Float16 sbuf[2][NK * LT];  // 2 x 10 KB
    const int c    = blockIdx.x;
    const int tid  = threadIdx.x;
    const int wave = tid >> 5;
    const int lane = tid & 31;
    const int lo16 = lane & 15;
    const int hiK  = (lane < 16) ? 0 : 8;   // ISA 16-bit fragment lane split

    v8f acc[8];
#pragma unroll
    for (int j = 0; j < 8; ++j) { v8f z = {}; acc[j] = z; }

    const int NITER = NB / 32;
    stage_chunk(feat, lab, c, 0, sbuf[0], tid);

    for (int it = 0; it < NITER; ++it) {
        _Float16* cur = sbuf[it & 1];
        _Float16* nxt = sbuf[(it & 1) ^ 1];
        __syncthreads();  // cur fully staged; nxt no longer being read
        if (it + 1 < NITER)
            stage_chunk(feat, lab, c, (it + 1) * 32, nxt, tid);

        v16h fa = frag(cur, wave * 16 + lo16, hiK);
#pragma unroll
        for (int j = 0; j < 8; ++j) {
            v16h fb = frag(cur, j * 16 + lo16, hiK);
            acc[j] = __builtin_amdgcn_wmma_f32_16x16x32_f16(
                false, fa, false, fb, (short)0, acc[j], false, false);
        }
    }

    // epilogue: S = R/n - mu mu^T + I
    int   n    = counts[c];
    float invn = (n > 0) ? 1.0f / (float)n : 0.0f;
    const float* muc = mu + c * NK;
    float* Sc = S + (size_t)c * NK * NK;
#pragma unroll
    for (int j = 0; j < 8; ++j) {
#pragma unroll
        for (int r = 0; r < 8; ++r) {
            int iloc = wave * 16 + r + hiK;   // D layout: VGPR r -> M=r / M=8+r
            int jloc = j * 16 + lo16;
            float v = acc[j][r] * invn - muc[iloc] * muc[jloc]
                    + ((iloc == jloc) ? 1.0f : 0.0f);
            Sc[(size_t)iloc * NK + jloc] = v;
        }
    }
}

// ---------------------------------------------------------------------------
// Gauss-Jordan inversion of SPD S_c (diag >= 1, pivot-free is stable);
// log2(det) = sum log2(pivots). grid = 32, block = 128. Augmented matrix in
// global scratch; pivot row/col cached in LDS each step.
// ---------------------------------------------------------------------------
__global__ void k_invert(const float* __restrict__ S, float* aug,
                         float* inv, float* ld2) {
    __shared__ float pivrow[256];
    __shared__ float pivcol[128];
    const int c = blockIdx.x;
    const int t = threadIdx.x;
    float* A = aug + (size_t)c * NK * 256;
    const float* Sc = S + (size_t)c * NK * NK;

    for (int r = 0; r < NK; ++r) {
        A[r * 256 + t]       = Sc[r * NK + t];
        A[r * 256 + 128 + t] = (r == t) ? 1.0f : 0.0f;
    }
    __threadfence_block();
    __syncthreads();

    float ld = 0.0f;
    for (int p = 0; p < NK; ++p) {
        float piv = A[p * 256 + p];
        float ip  = 1.0f / piv;
        pivrow[t]       = A[p * 256 + t] * ip;
        pivrow[t + 128] = A[p * 256 + 128 + t] * ip;
        pivcol[t]       = A[t * 256 + p];
        if (t == 0) ld += log2f(fabsf(piv));
        __syncthreads();
        for (int r = 0; r < NK; ++r) {
            if (r == p) {
                A[r * 256 + t]       = pivrow[t];
                A[r * 256 + 128 + t] = pivrow[t + 128];
            } else {
                float f = pivcol[r];
                A[r * 256 + t]       -= f * pivrow[t];
                A[r * 256 + 128 + t] -= f * pivrow[t + 128];
            }
        }
        __threadfence_block();
        __syncthreads();
    }
    for (int r = 0; r < NK; ++r)
        inv[(size_t)c * NK * NK + r * NK + t] = A[r * 256 + 128 + t];
    if (t == 0) ld2[c] = ld;
}

// ---------------------------------------------------------------------------
// KL pairs: quad = d^T inv_j d ; tr = sum_ab inv_j[a,b] S_i[b,a]
// grid = 1024 (pair = i*32+j), block = 128.
// ---------------------------------------------------------------------------
__global__ void k_kl(const float* __restrict__ mu, const float* __restrict__ S,
                     const float* __restrict__ inv, const float* __restrict__ ld2,
                     float* kl) {
    __shared__ float d[128];
    __shared__ float red[128];
    const int pair = blockIdx.x;
    const int i = pair >> 5, j = pair & 31;
    const int t = threadIdx.x;
    d[t] = mu[i * NK + t] - mu[j * NK + t];
    __syncthreads();
    const float* invj = inv + (size_t)j * NK * NK;
    const float* Si   = S   + (size_t)i * NK * NK;
    float v = 0.0f, trp = 0.0f;
    for (int b = 0; b < NK; ++b) v   += invj[t * NK + b] * d[b];
    for (int b = 0; b < NK; ++b) trp += invj[t * NK + b] * Si[b * NK + t];
    red[t] = v * d[t] + trp;
    __syncthreads();
    for (int s = 64; s > 0; s >>= 1) {
        if (t < s) red[t] += red[t + s];
        __syncthreads();
    }
    if (t == 0) kl[pair] = 0.5f * ((ld2[j] - ld2[i]) - (float)NK + red[0]);
}

// ---------------------------------------------------------------------------
// Appearance-rank mask + final reduction. grid = 1, block = 256.
// ---------------------------------------------------------------------------
__global__ void k_final(const int* __restrict__ first,
                        const float* __restrict__ kl,
                        const float* __restrict__ mu, float* out) {
    __shared__ int   rank[NC];
    __shared__ float red[256];
    const int t = threadIdx.x;
    if (t < NC) {
        int fc = first[t], rk = 0;
        for (int c2 = 0; c2 < NC; ++c2) {
            int f2 = first[c2];
            rk += (f2 < fc) || (f2 == fc && c2 < t);
        }
        rank[t] = rk;
    }
    __syncthreads();
    float s = 0.0f;
    for (int p = t; p < NC * NC; p += 256) {
        int i = p >> 5, j = p & 31;
        if (rank[i] <= NC - 2 && rank[j] >= 1) s += kl[p];
    }
    for (int e = t; e < NC * NK; e += 256) { float m = mu[e]; s -= m * m; }
    red[t] = s;
    __syncthreads();
    for (int sft = 128; sft > 0; sft >>= 1) {
        if (t < sft) red[t] += red[t + sft];
        __syncthreads();
    }
    if (t == 0) out[0] = red[0];
}

// ---------------------------------------------------------------------------
extern "C" void kernel_launch(void* const* d_in, const int* in_sizes, int n_in,
                              void* d_out, int out_size, void* d_ws, size_t ws_size,
                              hipStream_t stream) {
    const float* feat = (const float*)d_in[0];   // [B,K] f32
    const int*   lab  = (const int*)d_in[1];     // [B]   i32
    (void)in_sizes; (void)n_in; (void)out_size; (void)ws_size;

    float* ws     = (float*)d_ws;
    float* musum  = ws + OFF_MUSUM;
    int*   counts = (int*)(ws + OFF_COUNTS);
    int*   first  = (int*)(ws + OFF_FIRST);
    float* mu     = ws + OFF_MU;
    float* ld2    = ws + OFF_LD2;
    float* kl     = ws + OFF_KL;
    float* S      = ws + OFF_S;
    float* inv    = ws + OFF_INV;
    float* aug    = ws + OFF_AUG;
    float* out    = (float*)d_out;

    k_init    <<<dim3((NC * NK + 255) / 256), dim3(256), 0, stream>>>(musum, counts, first);
    k_classsum<<<dim3(NB / 256),              dim3(128), 0, stream>>>(feat, lab, musum, counts);
    k_first   <<<dim3(NB / 256),              dim3(256), 0, stream>>>(lab, first);
    k_mu      <<<dim3((NC * NK + 255) / 256), dim3(256), 0, stream>>>(musum, counts, mu);
    k_gram    <<<dim3(NC),                    dim3(256), 0, stream>>>(feat, lab, mu, counts, S);
    k_invert  <<<dim3(NC),                    dim3(128), 0, stream>>>(S, aug, inv, ld2);
    k_kl      <<<dim3(NC * NC),               dim3(128), 0, stream>>>(mu, S, inv, ld2, kl);
    k_final   <<<dim3(1),                     dim3(256), 0, stream>>>(first, kl, mu, out);
}
